// KernelProjectionT2I_55267639165043
// MI455X (gfx1250) — matrix operand order repaired and checked by reading, not compiled, not run on hardware
//
#include <hip/hip_runtime.h>
#include <hip/hip_bf16.h>

typedef __attribute__((ext_vector_type(16))) __bf16 v16bf;
typedef __attribute__((ext_vector_type(8)))  float  v8f;

#define QN 48
#define BN 48
#define DIM 1024
#define RR 36
#define TT 32
#define KCHUNK 512
#define NWAVES 16

__device__ __forceinline__ unsigned short f32_to_bf16(float f) {
    unsigned int u = __float_as_uint(f);
    u += 0x7FFFu + ((u >> 16) & 1u);
    return (unsigned short)(u >> 16);
}

// ---------------------------------------------------------------------------
// Kernel A: pack Wconv (1024x1024 f32, row-major [d][c]) into bf16 WMMA
// A-fragments: layout [mtile(64)][kstep(32)][lane(32)][elem(16)].
// Per ISA 16-bit A-matrix 16x32 table: M = lane&15,
// K = (i&7) | ((i&8)<<1) | ((lane&16)>>1)  within the 32-wide k-step.
// ---------------------------------------------------------------------------
__global__ __launch_bounds__(256) void pack_wconv(const float* __restrict__ Wconv,
                                                  unsigned short* __restrict__ Apack) {
    int g = blockIdx.x * 256 + threadIdx.x;   // 0..65535
    int mtile = g >> 10;
    int kstep = (g >> 5) & 31;
    int lane  = g & 31;
    int M = mtile * 16 + (lane & 15);
    int kadd = (lane & 16) >> 1;
    unsigned short* dst = Apack + (size_t)g * 16;
#pragma unroll
    for (int i = 0; i < 16; ++i) {
        int K = kstep * 32 + ((i & 7) | ((i & 8) << 1) | kadd);
        dst[i] = f32_to_bf16(Wconv[(size_t)M * DIM + K]);
    }
}

// ---------------------------------------------------------------------------
// Kernel B1: cap_repr[q][t] = dot(cap0[q], Wred[t]) + bred[t]   (48 x 256)
// plus inv_cap_norm[q] = rsqrt(sum(cap0^2)).
// ---------------------------------------------------------------------------
__global__ __launch_bounds__(256) void cap_reduce(const float* __restrict__ cap_embed,
                                                  const float* __restrict__ Wred,
                                                  const float* __restrict__ bred,
                                                  float* __restrict__ cap_repr,
                                                  float* __restrict__ inv_cap_norm) {
    __shared__ float red[8];
    int q = blockIdx.x, tid = threadIdx.x;
    const float* cap0 = cap_embed + (size_t)q * TT * DIM;   // cap_embed[q,0,:]
    const float* wr = Wred + (size_t)tid * DIM;
    float acc = 0.f;
    for (int d = 0; d < DIM; ++d) acc = fmaf(cap0[d], wr[d], acc);
    cap_repr[q * 256 + tid] = acc + bred[tid];

    float ss = 0.f;
    for (int d = tid; d < DIM; d += 256) { float c = cap0[d]; ss = fmaf(c, c, ss); }
    for (int off = 1; off < 32; off <<= 1) ss += __shfl_xor(ss, off, 32);
    if ((tid & 31) == 0) red[tid >> 5] = ss;
    __syncthreads();
    if (tid == 0) {
        float S = 0.f;
        for (int w = 0; w < 8; ++w) S += red[w];
        inv_cap_norm[q] = rsqrtf(S);
    }
}

// ---------------------------------------------------------------------------
// Kernel B2: wdyn_raw[q][o] = dot(cap_repr[q], Wproj[o]) + bproj[o]  (48 x 3072)
// ---------------------------------------------------------------------------
__global__ __launch_bounds__(256) void proj_kernel(const float* __restrict__ cap_repr,
                                                   const float* __restrict__ Wproj,
                                                   const float* __restrict__ bproj,
                                                   float* __restrict__ wdyn_raw) {
    int q = blockIdx.x;
    int o = blockIdx.y * 256 + threadIdx.x;   // 0..3071
    const float* cr = cap_repr + q * 256;
    const float* wp = Wproj + (size_t)o * 256;
    float acc = 0.f;
    for (int j = 0; j < 256; ++j) acc = fmaf(cr[j], wp[j], acc);
    wdyn_raw[(size_t)q * 3072 + o] = acc + bproj[o];
}

// ---------------------------------------------------------------------------
// Kernel B3: softmax over K=3, write as 3 planes [k][q][d] for coalesced use.
// ---------------------------------------------------------------------------
__global__ __launch_bounds__(256) void softmax3(const float* __restrict__ wdyn_raw,
                                                float* __restrict__ planes) {
    int q = blockIdx.x, tid = threadIdx.x;
    for (int d = tid; d < DIM; d += 256) {
        const float* src = wdyn_raw + (size_t)q * 3072 + d * 3;
        float a = src[0], b = src[1], c = src[2];
        float m = fmaxf(a, fmaxf(b, c));
        float ea = __expf(a - m), eb = __expf(b - m), ec = __expf(c - m);
        float inv = 1.f / (ea + eb + ec);
        planes[0 * QN * DIM + q * DIM + d] = ea * inv;
        planes[1 * QN * DIM + q * DIM + d] = eb * inv;
        planes[2 * QN * DIM + q * DIM + d] = ec * inv;
    }
}

// ---------------------------------------------------------------------------
// Kernel C: fused per-(q,b):
//   conv_dyn (dynamic 3-tap conv) -> LDS as bf16 B-fragments
//   y = Wconv @ conv_dyn  via v_wmma_f32_16x16x32_bf16 (M=1024,N=48pad,K=1024)
//   + bconv, softmax over r, pooled vec, l2-norm, dot with cap0
// ---------------------------------------------------------------------------
__global__ __launch_bounds__(512) void fused_gemm(const float* __restrict__ img,
                                                  const unsigned short* __restrict__ Apack,
                                                  const float* __restrict__ planes,
                                                  const float* __restrict__ cap_embed,
                                                  const float* __restrict__ inv_cap_norm,
                                                  const float* __restrict__ bconv,
                                                  float* __restrict__ out) {
    // B fragments: [kstep(16)][ntile(3)][lane(32)][elem(16)] bf16 = 48 KB
    __shared__ __align__(32) unsigned short Blds[16 * 3 * 32 * 16];
    __shared__ float vbuf[DIM];
    __shared__ float red[2 * NWAVES];

    const int q = blockIdx.x, b = blockIdx.y;
    const int tid = threadIdx.x;
    const int lane = tid & 31;
    const int wave = tid >> 5;             // 0..15

    const float* w0 = planes + 0 * QN * DIM + q * DIM;
    const float* w1 = planes + 1 * QN * DIM + q * DIM;
    const float* w2 = planes + 2 * QN * DIM + q * DIM;
    const float* imgb = img + (size_t)b * RR * DIM;

    v8f acc[4][3];
#pragma unroll
    for (int mt = 0; mt < 4; ++mt)
#pragma unroll
        for (int nt = 0; nt < 3; ++nt)
            acc[mt][nt] = (v8f){0.f, 0.f, 0.f, 0.f, 0.f, 0.f, 0.f, 0.f};

    for (int chunk = 0; chunk < 2; ++chunk) {
        const int k0 = chunk * KCHUNK;

        // zero LDS (covers the padded columns r=36..47)
        unsigned int* bz = (unsigned int*)Blds;
#pragma unroll
        for (int j = tid; j < 16 * 3 * 32 * 16 / 2; j += 512) bz[j] = 0u;
        __syncthreads();

        // build conv_dyn bf16 B-fragments: conv[c][r] = w0*x[r-1]+w1*x[r]+w2*x[r+1]
        for (int idx = tid; idx < KCHUNK * RR; idx += 512) {
            int r  = idx >> 9;          // 0..35
            int cl = idx & (KCHUNK - 1);
            int c  = k0 + cl;
            float xm = (r > 0)      ? imgb[(r - 1) * DIM + c] : 0.f;
            float xc =                imgb[r * DIM + c];
            float xp = (r < RR - 1) ? imgb[(r + 1) * DIM + c] : 0.f;
            float val = w0[c] * xm + w1[c] * xc + w2[c] * xp;
            int kstep = cl >> 5, kk = cl & 31;
            int ntile = r >> 4, nl = r & 15;
            int l2 = nl | ((kk & 8) << 1);
            int i  = (kk & 7) | ((kk & 16) >> 1);
            Blds[((((kstep * 3 + ntile) * 32) + l2) << 4) | i] = f32_to_bf16(val);
        }
        __syncthreads();

        // GEMM over this K chunk: each wave owns M rows [wave*64, wave*64+64)
        const v16bf* Bfrag = (const v16bf*)Blds;
        const v16bf* Afrag = (const v16bf*)Apack;
#pragma unroll
        for (int kstep = 0; kstep < 16; ++kstep) {
            v16bf bfr0 = Bfrag[(kstep * 3 + 0) * 32 + lane];
            v16bf bfr1 = Bfrag[(kstep * 3 + 1) * 32 + lane];
            v16bf bfr2 = Bfrag[(kstep * 3 + 2) * 32 + lane];
            int ks = chunk * 16 + kstep;
#pragma unroll
            for (int mt = 0; mt < 4; ++mt) {
                int mtile = wave * 4 + mt;
                v16bf af = Afrag[((size_t)mtile * 32 + ks) * 32 + lane];
                acc[mt][0] = __builtin_amdgcn_wmma_f32_16x16x32_bf16(
                    false, af, false, bfr0, (short)0, acc[mt][0], false, false);
                acc[mt][1] = __builtin_amdgcn_wmma_f32_16x16x32_bf16(
                    false, af, false, bfr1, (short)0, acc[mt][1], false, false);
                acc[mt][2] = __builtin_amdgcn_wmma_f32_16x16x32_bf16(
                    false, af, false, bfr2, (short)0, acc[mt][2], false, false);
            }
        }
        __syncthreads();
    }

    // --- per-row softmax over r (N dim) + weighted pooling -----------------
    // C-layout: VGPR v: lanes 0-15 -> M=v, N=lane; lanes 16-31 -> M=v+8, N=lane-16
    const int hl = (lane >> 4) & 1;
    const int nl = lane & 15;
    const bool valid2 = nl < (RR - 32);   // last N-tile: only r=32..35 valid
#pragma unroll
    for (int mt = 0; mt < 4; ++mt) {
#pragma unroll
        for (int v = 0; v < 8; ++v) {
            int m = wave * 64 + mt * 16 + v + hl * 8;
            float bc = bconv[m];
            float y0 = acc[mt][0][v] + bc;
            float y1 = acc[mt][1][v] + bc;
            float y2 = acc[mt][2][v] + bc;
            float mx = fmaxf(y0, y1);
            mx = fmaxf(mx, valid2 ? y2 : -3.4e38f);
            for (int off = 1; off < 16; off <<= 1) mx = fmaxf(mx, __shfl_xor(mx, off, 32));
            float e0 = __expf(y0 - mx), e1 = __expf(y1 - mx);
            float e2 = valid2 ? __expf(y2 - mx) : 0.f;
            float s = e0 + e1 + e2;
            float t = e0 * y0 + e1 * y1 + e2 * y2;
            for (int off = 1; off < 16; off <<= 1) {
                s += __shfl_xor(s, off, 32);
                t += __shfl_xor(t, off, 32);
            }
            if (nl == 0) vbuf[m] = t / s;
        }
    }
    __syncthreads();

    // --- l2-norm of pooled vec + dot with cap0 -----------------------------
    const float* cap0 = cap_embed + (size_t)q * TT * DIM;
    float ss = 0.f, dt = 0.f;
    for (int d = tid; d < DIM; d += 512) {
        float vv = vbuf[d];
        ss = fmaf(vv, vv, ss);
        dt = fmaf(vv, cap0[d], dt);
    }
    for (int off = 1; off < 32; off <<= 1) {
        ss += __shfl_xor(ss, off, 32);
        dt += __shfl_xor(dt, off, 32);
    }
    if (lane == 0) { red[wave] = ss; red[NWAVES + wave] = dt; }
    __syncthreads();
    if (tid == 0) {
        float S = 0.f, Dt = 0.f;
        for (int w = 0; w < NWAVES; ++w) { S += red[w]; Dt += red[NWAVES + w]; }
        out[b * QN + q] = Dt * rsqrtf(S) * inv_cap_norm[q];
    }
}

// ---------------------------------------------------------------------------
extern "C" void kernel_launch(void* const* d_in, const int* in_sizes, int n_in,
                              void* d_out, int out_size, void* d_ws, size_t ws_size,
                              hipStream_t stream) {
    const float* img   = (const float*)d_in[0];   // (48,36,1024)
    const float* cap   = (const float*)d_in[1];   // (48,32,1024)
    // d_in[2] = lens (unused by reference math)
    const float* Wred  = (const float*)d_in[3];   // (256,1024)
    const float* bred  = (const float*)d_in[4];   // (256,)
    const float* Wproj = (const float*)d_in[5];   // (3072,256)
    const float* bproj = (const float*)d_in[6];   // (3072,)
    const float* Wconv = (const float*)d_in[7];   // (1024,1024)
    const float* bconv = (const float*)d_in[8];   // (1024,)
    float* out = (float*)d_out;                   // (48,48)

    unsigned short* Apack = (unsigned short*)d_ws;          // 2 MB bf16 fragments
    float* fbase    = (float*)((char*)d_ws + 2u * 1024u * 1024u);
    float* cap_repr = fbase;                                // 48*256
    float* wdyn_raw = cap_repr + QN * 256;                  // 48*3072
    float* planes   = wdyn_raw + QN * 3072;                 // 3*48*1024
    float* icn      = planes + 3 * QN * DIM;                // 48

    pack_wconv<<<256, 256, 0, stream>>>(Wconv, Apack);
    cap_reduce<<<QN, 256, 0, stream>>>(cap, Wred, bred, cap_repr, icn);
    proj_kernel<<<dim3(QN, 12), 256, 0, stream>>>(cap_repr, Wproj, bproj, wdyn_raw);
    softmax3<<<QN, 256, 0, stream>>>(wdyn_raw, planes);
    fused_gemm<<<dim3(QN, BN), 512, 0, stream>>>(img, Apack, planes, cap, icn, bconv, out);
}
